// PreTrainModel_80487687127151
// MI455X (gfx1250) — matrix operand order repaired and checked by reading, not compiled, not run on hardware
//
#include <hip/hip_runtime.h>
#include <math.h>

// ---------------------------------------------------------------------------
// Types for CDNA5 WMMA (wave32)
// ---------------------------------------------------------------------------
typedef __bf16 bf16_t;
typedef __attribute__((ext_vector_type(16))) __bf16 v16bf;
typedef __attribute__((ext_vector_type(8)))  float  v8f;

// ---------------------------------------------------------------------------
// Problem constants (from reference)
// ---------------------------------------------------------------------------
constexpr int NUME    = 30000;
constexpr int DIM_IN  = 128;
constexpr int DIM_OUT = 128;
constexpr int DIM_R   = 64;
constexpr int DIM_T   = 32;
constexpr int DEEPTH  = 2;
constexpr int HEADS   = 4;
constexpr int DHEAD   = DIM_OUT / HEADS;   // 32
constexpr int R_TOTAL = 17;
constexpr int N_NODES = 20000;
constexpr int N_EDGES = 15000;
constexpr int BATCH   = 1024;
constexpr int DIN     = DIM_IN + DIM_T;    // 160
constexpr float NEG_SLOPE = 0.2f;

// GEMM tiling
constexpr int KT_X   = DIN / 32;           // 5 k-tiles for feature GEMM
constexpr int NT_Z   = DIM_OUT / 16;       // 8 n-tiles
constexpr int MT_X   = N_NODES / 16;       // 1250 m-tiles
constexpr int KDIM_C = DIM_OUT + DIM_R;    // 192
constexpr int KT_C   = KDIM_C / 32;        // 6
constexpr int N_CLS  = NUME;               // 30000
constexpr int N_CPAD = 30016;              // pad to 64 cols multiple (1876 = 4*469)
constexpr int NT_C   = N_CPAD / 16;        // 1876
constexpr int MT_C   = BATCH / 16;         // 64

// ---------------------------------------------------------------------------
// Fragment-packed layouts (ISA 7.12.2, wave32):
//  A tile 16(M)x32(K): lane = ((k%32)/8 % 2)*16 + m%16 ;
//                      elem = (k%8) + ((k%32)/16)*8  (two 8-elem K chunks)
//  B tile 32(K)x16(N): lane = ((k%32)/8 % 2)*16 + n%16 ; elem = (k%8) + ((k%32)/16)*8
// Each tile = 32 lanes * 16 bf16 = 512 elements, so a lane's fragment is one
// contiguous 32-byte v16bf load.
// ---------------------------------------------------------------------------
__device__ __forceinline__ size_t packA_off(int m, int k, int numKt) {
  int mt = m >> 4, kt = k >> 5;
  int r = m & 15, c = k & 31;
  int lane = (((c >> 3) & 1) << 4) + r;
  int e = (c & 7) + ((c >> 4) << 3);
  return (((size_t)(mt * numKt + kt)) << 9) + (size_t)lane * 16 + e;
}
__device__ __forceinline__ size_t packB_off(int k, int n, int numNt) {
  int kt = k >> 5, nt = n >> 4;
  int lane = (((k >> 3) & 1) << 4) + (n & 15);
  int e = (k & 7) + (((k >> 4) & 1) << 3);
  return (((size_t)(kt * numNt + nt)) << 9) + (size_t)lane * 16 + e;
}

// ---------------------------------------------------------------------------
// bf16 WMMA GEMM: one wave computes a (16*MBLK) x (16*NBLK) block of C.
// A: fragment-packed (Mt x Kt tiles), B: fragment-packed (Kt x Nt tiles).
// Requires Mt % MBLK == 0 and Nt % NBLK == 0 (exact in both uses).
// MBLK*NBLK WMMAs per k-step amortize each fragment load over MBLK (B) /
// NBLK (A) matrix ops; all guards are wave-uniform so EXEC stays all-1s.
// ---------------------------------------------------------------------------
template <int MBLK, int NBLK>
__global__ __launch_bounds__(256)
void k_gemm_bf16(const bf16_t* __restrict__ A, const bf16_t* __restrict__ B,
                 float* __restrict__ C, const float* __restrict__ bias,
                 int Mb, int Kt, int Nt, int ldc, int nValid) {
  const int gtid = blockIdx.x * blockDim.x + threadIdx.x;
  const int wave = gtid >> 5;
  const int lane = threadIdx.x & 31;
  const int nBlocks = Nt / NBLK;
  if (wave >= Mb * nBlocks) return;           // wave-uniform early exit
  const int mt0 = (wave / nBlocks) * MBLK;
  const int nt0 = (wave % nBlocks) * NBLK;

  v8f acc[MBLK][NBLK] = {};

  for (int kt = 0; kt < Kt; ++kt) {
    v16bf a[MBLK];
#pragma unroll
    for (int i = 0; i < MBLK; ++i)
      a[i] = *(const v16bf*)(A + (((size_t)(mt0 + i) * Kt + kt) << 9) +
                             (size_t)lane * 16);
#pragma unroll
    for (int j = 0; j < NBLK; ++j) {
      v16bf b = *(const v16bf*)(B + (((size_t)kt * Nt + nt0 + j) << 9) +
                                (size_t)lane * 16);
#pragma unroll
      for (int i = 0; i < MBLK; ++i)
        acc[i][j] = __builtin_amdgcn_wmma_f32_16x16x32_bf16(
            /*neg_a=*/false, a[i], /*neg_b=*/false, b,
            /*c_mod=*/(short)0, acc[i][j], /*reuse_a=*/false, /*reuse_b=*/false);
    }
  }

  // C/D layout: lane L -> col = L%16, rows = (L/16)*8 + v
  const int rsub = (lane >> 4) << 3;
  const int colL = lane & 15;
#pragma unroll
  for (int i = 0; i < MBLK; ++i) {
    const int row0 = (mt0 + i) * 16 + rsub;
#pragma unroll
    for (int j = 0; j < NBLK; ++j) {
      int col = (nt0 + j) * 16 + colL;
      if (col >= nValid) continue;
      float badd = bias ? bias[col] : 0.0f;
#pragma unroll
      for (int v = 0; v < 8; ++v)
        C[(size_t)(row0 + v) * ldc + col] = acc[i][j][v] + badd;
    }
  }
}

// ---------------------------------------------------------------------------
// Elementwise / graph kernels
// ---------------------------------------------------------------------------
__global__ void k_build_x0(const int* __restrict__ node_ids,
                           const float* __restrict__ entity_emb,
                           const float* __restrict__ basis_freq,
                           const float* __restrict__ phase,
                           const int* __restrict__ ts,
                           bf16_t* __restrict__ packX,
                           float* __restrict__ phi) {
  int idx = blockIdx.x * blockDim.x + threadIdx.x;
  if (idx >= N_NODES * DIN) return;
  int n = idx / DIN, c = idx % DIN;
  int id = node_ids[n];
  float v;
  if (c < DIM_IN) {
    v = entity_emb[(size_t)(id % NUME) * DIM_IN + c];
  } else {
    int ct = c - DIM_IN;
    float dt = (float)((ts[0] - 1) - id / NUME);
    v = cosf(dt * basis_freq[ct] + phase[ct]);
    phi[n * DIM_T + ct] = v;
  }
  packX[packA_off(n, c, KT_X)] = (bf16_t)v;
}

__global__ void k_pack_fcw(const float* __restrict__ fc_w, bf16_t* __restrict__ packW) {
  int idx = blockIdx.x * blockDim.x + threadIdx.x;
  int total = DEEPTH * R_TOTAL * DIN * DIM_OUT;
  if (idx >= total) return;
  int mat = idx / (DIN * DIM_OUT);
  int rem = idx % (DIN * DIM_OUT);
  int k = rem / DIM_OUT, n = rem % DIM_OUT;
  packW[(size_t)mat * (KT_X * NT_Z * 512) + packB_off(k, n, NT_Z)] = (bf16_t)fc_w[idx];
}

__global__ void k_pack_clsw(const float* __restrict__ subW, const float* __restrict__ objW,
                            bf16_t* __restrict__ packSub, bf16_t* __restrict__ packObj) {
  int idx = blockIdx.x * blockDim.x + threadIdx.x;
  int per = KDIM_C * N_CPAD;
  if (idx >= 2 * per) return;
  int which = idx / per;
  int rem = idx % per;
  int k = rem / N_CPAD, n = rem % N_CPAD;
  const float* W = which ? objW : subW;
  float v = (n < N_CLS) ? W[(size_t)k * N_CLS + n] : 0.0f;
  bf16_t* dst = which ? packObj : packSub;
  dst[packB_off(k, n, NT_C)] = (bf16_t)v;
}

__global__ void k_zero_f32(float* __restrict__ p, int n) {
  int idx = blockIdx.x * blockDim.x + threadIdx.x;
  if (idx < n) p[idx] = 0.0f;
}

__global__ void k_attn_lr(const float* __restrict__ z, const float* __restrict__ al,
                          const float* __restrict__ ar,
                          float* __restrict__ el, float* __restrict__ er) {
  int idx = blockIdx.x * blockDim.x + threadIdx.x;
  if (idx >= N_NODES * HEADS) return;
  int n = idx >> 2, h = idx & 3;
  const float* zp = z + (size_t)n * DIM_OUT + h * DHEAD;
  float sl = 0.f, sr = 0.f;
#pragma unroll
  for (int d = 0; d < DHEAD; ++d) {
    float zv = zp[d];
    sl += zv * al[h * DHEAD + d];
    sr += zv * ar[h * DHEAD + d];
  }
  el[idx] = sl;
  er[idx] = sr;
}

__global__ void k_init_md(float* __restrict__ m, float* __restrict__ den) {
  int idx = blockIdx.x * blockDim.x + threadIdx.x;
  if (idx >= N_NODES * HEADS) return;
  m[idx] = -1e30f;
  den[idx] = 0.0f;
}

__device__ __forceinline__ void atomicMaxF(float* addr, float val) {
  // sign-aware float max via integer atomics
  if (val >= 0.0f) atomicMax((int*)addr, __float_as_int(val));
  else             atomicMin((unsigned int*)addr, __float_as_uint(val));
}

__device__ __forceinline__ float edge_logit(const float* el, const float* er,
                                            int s, int t, int h) {
  float e = el[s * HEADS + h] + er[t * HEADS + h];
  return (e >= 0.0f) ? e : NEG_SLOPE * e;
}

__global__ void k_edge_max(const int* __restrict__ src, const int* __restrict__ dst,
                           const float* __restrict__ el, const float* __restrict__ er,
                           float* __restrict__ m) {
  int idx = blockIdx.x * blockDim.x + threadIdx.x;
  if (idx >= N_EDGES * HEADS) return;
  int e = idx >> 2, h = idx & 3;
  int s = src[e], t = dst[e];
  atomicMaxF(&m[t * HEADS + h], edge_logit(el, er, s, t, h));
}

__global__ void k_fix_m(float* __restrict__ m) {
  int idx = blockIdx.x * blockDim.x + threadIdx.x;
  if (idx >= N_NODES * HEADS) return;
  float v = m[idx];
  m[idx] = (v <= -1e29f) ? 0.0f : v;   // segment_max empty -> -inf -> 0
}

__global__ void k_edge_den(const int* __restrict__ src, const int* __restrict__ dst,
                           const float* __restrict__ el, const float* __restrict__ er,
                           const float* __restrict__ m, float* __restrict__ den) {
  int idx = blockIdx.x * blockDim.x + threadIdx.x;
  if (idx >= N_EDGES * HEADS) return;
  int e = idx >> 2, h = idx & 3;
  int s = src[e], t = dst[e];
  float lg = edge_logit(el, er, s, t, h);
  atomicAdd(&den[t * HEADS + h], expf(lg - m[t * HEADS + h]));
}

__global__ void k_edge_msg(const int* __restrict__ src, const int* __restrict__ dst,
                           const float* __restrict__ el, const float* __restrict__ er,
                           const float* __restrict__ m, const float* __restrict__ den,
                           const float* __restrict__ z, float* __restrict__ hacc) {
  int idx = blockIdx.x * blockDim.x + threadIdx.x;
  if (idx >= N_EDGES * DIM_OUT) return;
  int e = idx >> 7, f = idx & 127;
  int h = f >> 5;
  int s = src[e], t = dst[e];
  float lg = edge_logit(el, er, s, t, h);
  float ex = expf(lg - m[t * HEADS + h]);
  float alpha = ex / fmaxf(den[t * HEADS + h], 1e-9f);
  atomicAdd(&hacc[(size_t)t * DIM_OUT + f], alpha * z[(size_t)s * DIM_OUT + f]);
}

__global__ void k_finalize(const float* __restrict__ hacc, const float* __restrict__ phi,
                           float* __restrict__ hout, bf16_t* __restrict__ packX,
                           int writeH, int writePX) {
  int idx = blockIdx.x * blockDim.x + threadIdx.x;
  if (idx >= N_NODES * DIN) return;
  int n = idx / DIN, c = idx % DIN;
  if (c < DIM_OUT) {
    float v = fmaxf(hacc[(size_t)n * DIM_OUT + c] * (1.0f / (float)R_TOTAL), 0.0f);
    if (writeH)  hout[(size_t)n * DIM_OUT + c] = v;
    if (writePX) packX[packA_off(n, c, KT_X)] = (bf16_t)v;
  } else if (writePX) {
    packX[packA_off(n, c, KT_X)] = (bf16_t)phi[n * DIM_T + (c - DIM_OUT)];
  }
}

__global__ void k_build_clsA(const float* __restrict__ h,
                             const int* __restrict__ root_idx, const int* __restrict__ rel,
                             const float* __restrict__ sub_rel, const float* __restrict__ obj_rel,
                             bf16_t* __restrict__ packSubA, bf16_t* __restrict__ packObjA) {
  int idx = blockIdx.x * blockDim.x + threadIdx.x;
  int per = BATCH * KDIM_C;
  if (idx >= 2 * per) return;
  int which = idx / per;                 // 0: sub_pred A (obj side), 1: obj_pred A (sub side)
  int rem = idx % per;
  int b = rem / KDIM_C, c = rem % KDIM_C;
  int node = (which == 0) ? root_idx[BATCH + b] : root_idx[b];
  const float* relE = (which == 0) ? obj_rel : sub_rel;
  float v = (c < DIM_OUT) ? h[(size_t)node * DIM_OUT + c]
                          : relE[(size_t)rel[b] * DIM_R + (c - DIM_OUT)];
  bf16_t* dst = (which == 0) ? packSubA : packObjA;
  dst[packA_off(b, c, KT_C)] = (bf16_t)v;
}

// ---------------------------------------------------------------------------
// Host driver
// ---------------------------------------------------------------------------
static inline long cdivl(long a, long b) { return (a + b - 1) / b; }

extern "C" void kernel_launch(void* const* d_in, const int* in_sizes, int n_in,
                              void* d_out, int out_size, void* d_ws, size_t ws_size,
                              hipStream_t stream) {
  (void)in_sizes; (void)n_in; (void)out_size; (void)ws_size;

  const int*   node_ids   = (const int*)d_in[0];
  const int*   edge_src   = (const int*)d_in[1];   // (R_TOTAL, N_EDGES)
  const int*   edge_dst   = (const int*)d_in[2];
  const int*   root_idx   = (const int*)d_in[3];
  const int*   rel        = (const int*)d_in[4];
  const float* entity_emb = (const float*)d_in[5];
  const float* basis_freq = (const float*)d_in[6];
  const float* phase      = (const float*)d_in[7];
  const float* fc_w       = (const float*)d_in[8];
  const float* attn_l     = (const float*)d_in[9];
  const float* attn_r     = (const float*)d_in[10];
  const float* sub_rel    = (const float*)d_in[11];
  const float* obj_rel    = (const float*)d_in[12];
  const float* sub_cls_w  = (const float*)d_in[13];
  const float* sub_cls_b  = (const float*)d_in[14];
  const float* obj_cls_w  = (const float*)d_in[15];
  const float* obj_cls_b  = (const float*)d_in[16];
  const int*   ts         = (const int*)d_in[17];
  float*       out        = (float*)d_out;

  // ---- workspace bump allocator (256B aligned) ----
  size_t off = 0;
  auto alloc = [&](size_t bytes) -> void* {
    void* p = (char*)d_ws + off;
    off += (bytes + 255) & ~(size_t)255;
    return p;
  };
  bf16_t* packX    = (bf16_t*)alloc((size_t)MT_X * KT_X * 512 * sizeof(bf16_t));
  float*  phi      = (float*) alloc((size_t)N_NODES * DIM_T * sizeof(float));
  bf16_t* packFcW  = (bf16_t*)alloc((size_t)DEEPTH * R_TOTAL * KT_X * NT_Z * 512 * sizeof(bf16_t));
  float*  z        = (float*) alloc((size_t)N_NODES * DIM_OUT * sizeof(float));
  float*  hacc     = (float*) alloc((size_t)N_NODES * DIM_OUT * sizeof(float));
  float*  hbuf     = (float*) alloc((size_t)N_NODES * DIM_OUT * sizeof(float));
  float*  el       = (float*) alloc((size_t)N_NODES * HEADS * sizeof(float));
  float*  er       = (float*) alloc((size_t)N_NODES * HEADS * sizeof(float));
  float*  mmax     = (float*) alloc((size_t)N_NODES * HEADS * sizeof(float));
  float*  den      = (float*) alloc((size_t)N_NODES * HEADS * sizeof(float));
  bf16_t* packSubA = (bf16_t*)alloc((size_t)MT_C * KT_C * 512 * sizeof(bf16_t));
  bf16_t* packObjA = (bf16_t*)alloc((size_t)MT_C * KT_C * 512 * sizeof(bf16_t));
  bf16_t* packSubW = (bf16_t*)alloc((size_t)KT_C * NT_C * 512 * sizeof(bf16_t));
  bf16_t* packObjW = (bf16_t*)alloc((size_t)KT_C * NT_C * 512 * sizeof(bf16_t));

  const int TB = 256;

  // ---- stage inputs into bf16 fragment-packed buffers ----
  k_build_x0<<<cdivl((long)N_NODES * DIN, TB), TB, 0, stream>>>(
      node_ids, entity_emb, basis_freq, phase, ts, packX, phi);
  k_pack_fcw<<<cdivl((long)DEEPTH * R_TOTAL * DIN * DIM_OUT, TB), TB, 0, stream>>>(
      fc_w, packFcW);
  k_pack_clsw<<<cdivl(2L * KDIM_C * N_CPAD, TB), TB, 0, stream>>>(
      sub_cls_w, obj_cls_w, packSubW, packObjW);

  // ---- GAT layers ----
  for (int l = 0; l < DEEPTH; ++l) {
    k_zero_f32<<<cdivl((long)N_NODES * DIM_OUT, TB), TB, 0, stream>>>(hacc, N_NODES * DIM_OUT);
    for (int r = 0; r < R_TOTAL; ++r) {
      const bf16_t* Wr = packFcW + (size_t)(l * R_TOTAL + r) * (KT_X * NT_Z * 512);
      // z = x @ W_r via bf16 WMMA, f32 accumulate. MBLK=1 keeps 2500 waves in
      // flight for this small GEMM (latency hiding beats B-reuse here; B is
      // only 40KB and L2-resident).
      long waves = (long)MT_X * (NT_Z / 4);
      k_gemm_bf16<1, 4><<<cdivl(waves * 32, TB), TB, 0, stream>>>(
          packX, Wr, z, nullptr, MT_X, KT_X, NT_Z, DIM_OUT, DIM_OUT);
      const float* al = attn_l + (size_t)(l * R_TOTAL + r) * HEADS * DHEAD;
      const float* ar = attn_r + (size_t)(l * R_TOTAL + r) * HEADS * DHEAD;
      k_attn_lr<<<cdivl((long)N_NODES * HEADS, TB), TB, 0, stream>>>(z, al, ar, el, er);

      const int* src = edge_src + (size_t)r * N_EDGES;
      const int* dst = edge_dst + (size_t)r * N_EDGES;
      k_init_md<<<cdivl((long)N_NODES * HEADS, TB), TB, 0, stream>>>(mmax, den);
      k_edge_max<<<cdivl((long)N_EDGES * HEADS, TB), TB, 0, stream>>>(src, dst, el, er, mmax);
      k_fix_m<<<cdivl((long)N_NODES * HEADS, TB), TB, 0, stream>>>(mmax);
      k_edge_den<<<cdivl((long)N_EDGES * HEADS, TB), TB, 0, stream>>>(src, dst, el, er, mmax, den);
      k_edge_msg<<<cdivl((long)N_EDGES * DIM_OUT, TB), TB, 0, stream>>>(
          src, dst, el, er, mmax, den, z, hacc);
    }
    // h = relu(mean over relations); rebuild packed x for next layer or f32 h
    k_finalize<<<cdivl((long)N_NODES * DIN, TB), TB, 0, stream>>>(
        hacc, phi, hbuf, packX, /*writeH=*/(l == DEEPTH - 1) ? 1 : 0,
        /*writePX=*/(l == DEEPTH - 1) ? 0 : 1);
  }

  // ---- classifier heads ----
  k_build_clsA<<<cdivl(2L * BATCH * KDIM_C, TB), TB, 0, stream>>>(
      hbuf, root_idx, rel, sub_rel, obj_rel, packSubA, packObjA);

  // 64x64 per wave (4x4 C tiles): 16 WMMAs per fragment set, 4x less B
  // traffic from L2; 16*469 = 7504 waves, output-write (245MB) bound.
  long clsWaves = (long)(MT_C / 4) * (NT_C / 4);
  // sub_pred = [obj_emb | obj_rel_emb[rel]] @ sub_cls_w + sub_cls_b
  k_gemm_bf16<4, 4><<<cdivl(clsWaves * 32, TB), TB, 0, stream>>>(
      packSubA, packSubW, out, sub_cls_b, MT_C / 4, KT_C, NT_C, N_CLS, N_CLS);
  // obj_pred = [sub_emb | sub_rel_emb[rel]] @ obj_cls_w + obj_cls_b
  k_gemm_bf16<4, 4><<<cdivl(clsWaves * 32, TB), TB, 0, stream>>>(
      packObjA, packObjW, out + (size_t)BATCH * N_CLS, obj_cls_b, MT_C / 4, KT_C, NT_C, N_CLS, N_CLS);
}
